// GCN_3418793968209
// MI455X (gfx1250) — matrix operand order repaired and checked by reading, not compiled, not run on hardware
//
#include <hip/hip_runtime.h>
#include <hip/hip_bf16.h>
#include <cstdint>

// ---------------------------------------------------------------------------
// GCN forward for MI455X (gfx1250, wave32, WMMA).
// Dominant cost: two A@(XW) products, A = 16384x16384 fp32 (1 GiB each pass).
// Bandwidth floor: 2 GiB / 23.3 TB/s ~= 92 us. We use split-bf16 (hi/lo)
// v_wmma_f32_16x16x32_bf16 with fp32 accumulation: 3 WMMAs per 16x16x32 tile
// (Ah*Bh + Al*Bh + Ah*Bl) for near-fp32 accuracy at bf16 matrix throughput.
// Y (B-matrix) tiles are pre-split bf16 in global (L2-resident) and copied
// into LDS with the CDNA5 async engine (global_load_async_to_lds_b128,
// ASYNCcnt); A is streamed through registers for the VALU hi/lo split and
// prefetched ahead with global_prefetch_b8.
// ---------------------------------------------------------------------------

typedef __attribute__((ext_vector_type(16))) __bf16 bfx16;
typedef __attribute__((ext_vector_type(8)))  __bf16 v8bf;
typedef __attribute__((ext_vector_type(8)))  float  v8f;
typedef __attribute__((ext_vector_type(4)))  float  v4f;

constexpr int NTOT = 16384;   // nodes (= M = K of big GEMM)
constexpr int NEMB = 64;      // embedding width (= N of big GEMM)
constexpr int KT   = 32;      // K per WMMA step
constexpr int BM   = 64;      // rows per block in big GEMM
constexpr int LDA  = 40;      // padded LDS row stride in bf16 elems (32+8)
constexpr int NSTEP = NTOT / KT; // 512

__device__ inline bfx16 cat16(v8bf a, v8bf b) {
  return __builtin_shufflevector(a, b, 0,1,2,3,4,5,6,7,8,9,10,11,12,13,14,15);
}

__device__ inline v8f wmma_bf16(bfx16 a, bfx16 b, v8f c) {
  return __builtin_amdgcn_wmma_f32_16x16x32_bf16(
      /*neg_a=*/false, a, /*neg_b=*/false, b,
      /*c_mod=*/(short)0, c, /*reuse_a=*/false, /*reuse_b=*/false);
}

// CDNA5 async copy: 16B global -> LDS, tracked by ASYNCcnt (no VGPR return).
__device__ inline void async_copy16(uint32_t lds_off, const __bf16* g) {
  asm volatile("global_load_async_to_lds_b128 %0, %1, off"
               :
               : "v"(lds_off), "v"((uint64_t)(uintptr_t)g)
               : "memory");
}
__device__ inline void wait_async0() {
  asm volatile("s_wait_asynccnt 0" ::: "memory");
}

// Split 8 fp32 into bf16 hi + bf16 lo residual, store 16B each to LDS.
__device__ inline void split_store8(__bf16* dhi, __bf16* dlo, v4f a0, v4f a1) {
  v8bf hi, lo;
#pragma unroll
  for (int i = 0; i < 4; ++i) {
    float f = a0[i]; __bf16 h = (__bf16)f;
    hi[i] = h; lo[i] = (__bf16)(f - (float)h);
    float g = a1[i]; __bf16 h2 = (__bf16)g;
    hi[4 + i] = h2; lo[4 + i] = (__bf16)(g - (float)h2);
  }
  *(v8bf*)dhi = hi;
  *(v8bf*)dlo = lo;
}

// ---------------------------------------------------------------------------
// K0: embed  X0 = X @ We + be   ([16384,92] @ [92,64])
// ---------------------------------------------------------------------------
__global__ __launch_bounds__(256) void embed_kernel(
    const float* __restrict__ X, const float* __restrict__ We,
    const float* __restrict__ be, float* __restrict__ X0) {
  __shared__ float Xs[4][92];
  const int tid = threadIdx.x;
  const int rowBase = blockIdx.x * 4;
  for (int i = tid; i < 4 * 92; i += 256)
    Xs[i / 92][i % 92] = X[(size_t)rowBase * 92 + i];
  __syncthreads();
  const int r = tid >> 6, c = tid & 63;
  float acc = be[c];
  for (int k = 0; k < 92; ++k) acc += Xs[r][k] * We[k * 64 + c];
  X0[(size_t)(rowBase + r) * NEMB + c] = acc;
}

// ---------------------------------------------------------------------------
// K1: Y = Xcur @ W  (64x64), stored TRANSPOSED and pre-split into bf16 hi/lo:
//     Yt_hi/Yt_lo are [64][16384] bf16 (k-contiguous) for the WMMA B path.
// ---------------------------------------------------------------------------
__global__ __launch_bounds__(256) void xw_kernel(
    const float* __restrict__ Xc, const float* __restrict__ W,
    __bf16* __restrict__ Yh, __bf16* __restrict__ Yl) {
  __shared__ float Xs[64][65];
  const int tid = threadIdx.x;
  const int rowBase = blockIdx.x * 64;
#pragma unroll
  for (int i = 0; i < 16; ++i) {
    int lin = i * 256 + tid;
    Xs[lin >> 6][lin & 63] = Xc[(size_t)rowBase * NEMB + lin];
  }
  __syncthreads();
  const int col = tid >> 2;
  const int r0 = (tid & 3) * 16;
  float acc[16];
#pragma unroll
  for (int i = 0; i < 16; ++i) acc[i] = 0.f;
  for (int k = 0; k < 64; ++k) {
    float w = W[k * 64 + col];
#pragma unroll
    for (int i = 0; i < 16; ++i) acc[i] += Xs[r0 + i][k] * w;
  }
  v8bf h0, h1, l0, l1;
#pragma unroll
  for (int i = 0; i < 8; ++i) {
    float f = acc[i];     __bf16 h = (__bf16)f;
    h0[i] = h; l0[i] = (__bf16)(f - (float)h);
    float g = acc[8 + i]; __bf16 h2 = (__bf16)g;
    h1[i] = h2; l1[i] = (__bf16)(g - (float)h2);
  }
  const size_t base = (size_t)col * NTOT + rowBase + r0;
  *(v8bf*)(Yh + base) = h0; *(v8bf*)(Yh + base + 8) = h1;
  *(v8bf*)(Yl + base) = l0; *(v8bf*)(Yl + base + 8) = l1;
}

// ---------------------------------------------------------------------------
// K2: big GEMM + fused GC epilogue:
//     Xn = Xc + leaky_relu(A @ Y + b, 0.01)
// A fp32 streamed via registers (VALU hi/lo split); Y hi/lo copied to LDS by
// the async engine. Block: 256 threads = 8 waves; 64x64 output tile; K loop
// in steps of 32, double-buffered LDS; 6 WMMAs per wave per step.
// ---------------------------------------------------------------------------
__global__ __launch_bounds__(256) void spmm_kernel(
    const float* __restrict__ A, const __bf16* __restrict__ Bh,
    const __bf16* __restrict__ Bl, const float* __restrict__ Xc,
    const float* __restrict__ bias, float* __restrict__ Xn) {
  __shared__ __align__(16) __bf16 sAh[2][BM * LDA];
  __shared__ __align__(16) __bf16 sAl[2][BM * LDA];
  __shared__ __align__(16) __bf16 sBh[2][NEMB * LDA];
  __shared__ __align__(16) __bf16 sBl[2][NEMB * LDA];

  const int tid = threadIdx.x;
  const int rowBase = blockIdx.x * BM;

  // cooperative loader mapping: 64 rows (or n) x 32 k, 8 elems per thread
  const int lrow = tid >> 2;            // 0..63
  const int kseg = (tid & 3) * 8;       // 0,8,16,24
  const int ldsoff = lrow * LDA + kseg; // bf16 elements

  // wave/tile mapping
  const int lane  = tid & 31;
  const int wave  = tid >> 5;
  const int rt    = wave & 3;           // row tile 0..3
  const int ct0   = (wave >> 2) * 2;    // col tiles ct0, ct0+1
  const int lhalf = lane >> 4;          // 0/1
  const int l15   = lane & 15;

  const float*  aPtr  = A  + (size_t)(rowBase + lrow) * NTOT + kseg;
  const __bf16* bhPtr = Bh + (size_t)lrow * NTOT + kseg;
  const __bf16* blPtr = Bl + (size_t)lrow * NTOT + kseg;

  v8f acc0 = {0.f,0.f,0.f,0.f,0.f,0.f,0.f,0.f};
  v8f acc1 = {0.f,0.f,0.f,0.f,0.f,0.f,0.f,0.f};

  // prologue: stage 0 (Y via async engine, A via registers + split)
  async_copy16((uint32_t)(uintptr_t)&sBh[0][ldsoff], bhPtr);
  async_copy16((uint32_t)(uintptr_t)&sBl[0][ldsoff], blPtr);
  v4f a0 = *(const v4f*)(aPtr);
  v4f a1 = *(const v4f*)(aPtr + 4);
  split_store8(&sAh[0][ldsoff], &sAl[0][ldsoff], a0, a1);
  wait_async0();
  __syncthreads();

  for (int s = 0; s < NSTEP; ++s) {
    const int buf = s & 1;
    const int nb = buf ^ 1;
    if (s + 1 < NSTEP) {                 // prefetch next stage
      const int kB = (s + 1) * KT;
      async_copy16((uint32_t)(uintptr_t)&sBh[nb][ldsoff], bhPtr + kB);
      async_copy16((uint32_t)(uintptr_t)&sBl[nb][ldsoff], blPtr + kB);
      a0 = *(const v4f*)(aPtr + kB);
      a1 = *(const v4f*)(aPtr + kB + 4);
      // keep the 1 GiB HBM stream of A ahead of compute (L2 prefetch)
      __builtin_prefetch(aPtr + kB + 8 * KT, 0, 1);
    }
    // per-lane WMMA operand gathers (contiguous k = 16*half + s, same order for A and B), all 16B ds loads
    {
      const int arow = (rt * 16 + l15) * LDA;
      const int ak   = lhalf * 16;       // lanes>=16 hold K=16..31
      bfx16 Ah = cat16(*(const v8bf*)&sAh[buf][arow + ak],
                       *(const v8bf*)&sAh[buf][arow + ak + 8]);
      bfx16 Al = cat16(*(const v8bf*)&sAl[buf][arow + ak],
                       *(const v8bf*)&sAl[buf][arow + ak + 8]);
      const int bk = lhalf * 16;         // lanes>=16 hold K=16..31
      const int n0 = (ct0 * 16 + l15) * LDA;
      const int n1 = n0 + 16 * LDA;
      bfx16 B0h = cat16(*(const v8bf*)&sBh[buf][n0 + bk],
                        *(const v8bf*)&sBh[buf][n0 + bk + 8]);
      bfx16 B0l = cat16(*(const v8bf*)&sBl[buf][n0 + bk],
                        *(const v8bf*)&sBl[buf][n0 + bk + 8]);
      bfx16 B1h = cat16(*(const v8bf*)&sBh[buf][n1 + bk],
                        *(const v8bf*)&sBh[buf][n1 + bk + 8]);
      bfx16 B1l = cat16(*(const v8bf*)&sBl[buf][n1 + bk],
                        *(const v8bf*)&sBl[buf][n1 + bk + 8]);
      // split-precision product: Ah*Bh + Al*Bh + Ah*Bl (Al*Bl ~ eps^2, dropped)
      acc0 = wmma_bf16(Ah, B0h, acc0);
      acc0 = wmma_bf16(Al, B0h, acc0);
      acc0 = wmma_bf16(Ah, B0l, acc0);
      acc1 = wmma_bf16(Ah, B1h, acc1);
      acc1 = wmma_bf16(Al, B1h, acc1);
      acc1 = wmma_bf16(Ah, B1l, acc1);
    }
    if (s + 1 < NSTEP) {                 // fill the other buffer (A split)
      split_store8(&sAh[nb][ldsoff], &sAl[nb][ldsoff], a0, a1);
    }
    wait_async0();                       // async Y tiles landed in LDS
    __syncthreads();
  }

  // fused epilogue: Xn = Xc + leaky_relu(acc + bias)
#pragma unroll
  for (int t = 0; t < 2; ++t) {
    const v8f acc = t ? acc1 : acc0;
    const int col = (ct0 + t) * 16 + l15;
    const float bv = bias[col];
#pragma unroll
    for (int i = 0; i < 8; ++i) {
      const int row = rowBase + rt * 16 + lhalf * 8 + i;
      const float z = acc[i] + bv;
      const float lr = z > 0.f ? z : 0.01f * z;
      Xn[(size_t)row * NEMB + col] = Xc[(size_t)row * NEMB + col] + lr;
    }
  }
}

// ---------------------------------------------------------------------------
// K3a/K3b: deterministic two-pass BatchNorm statistics (no fp atomics)
// ---------------------------------------------------------------------------
__global__ __launch_bounds__(256) void bn_partial(
    const float* __restrict__ X, float* __restrict__ psum,
    float* __restrict__ psq) {
  const int tid = threadIdx.x;
  const int c = tid & 63, rl = tid >> 6;
  const int rowBase = blockIdx.x * 256;
  float s = 0.f, q = 0.f;
  for (int i = 0; i < 64; ++i) {
    float v = X[(size_t)(rowBase + rl + 4 * i) * NEMB + c];
    s += v; q += v * v;
  }
  __shared__ float ls[4][64], lq[4][64];
  ls[rl][c] = s; lq[rl][c] = q;
  __syncthreads();
  if (tid < 64) {
    psum[blockIdx.x * 64 + tid] =
        ls[0][tid] + ls[1][tid] + ls[2][tid] + ls[3][tid];
    psq[blockIdx.x * 64 + tid] =
        lq[0][tid] + lq[1][tid] + lq[2][tid] + lq[3][tid];
  }
}

__global__ void bn_final(const float* __restrict__ psum,
                         const float* __restrict__ psq,
                         const float* __restrict__ gamma,
                         const float* __restrict__ beta,
                         float* __restrict__ scale, float* __restrict__ shift) {
  const int c = threadIdx.x;  // 64 threads
  float S = 0.f, Q = 0.f;
  for (int b = 0; b < 64; ++b) { S += psum[b * 64 + c]; Q += psq[b * 64 + c]; }
  const float inv_n = 1.f / (float)NTOT;
  const float mean = S * inv_n;
  const float var = Q * inv_n - mean * mean;  // biased var
  const float sc = gamma[c] * rsqrtf(var + 1e-5f);
  scale[c] = sc;
  shift[c] = beta[c] - mean * sc;
}

// ---------------------------------------------------------------------------
// K4: fused BN-apply + ragged segment-mean pooling (one block per graph)
// ---------------------------------------------------------------------------
__global__ __launch_bounds__(256) void pool_kernel(
    const float* __restrict__ X, const int* __restrict__ Nseg,
    const float* __restrict__ scale, const float* __restrict__ shift,
    float* __restrict__ pooled) {
  const int g = blockIdx.x;
  const int tid = threadIdx.x;
  const int c = tid & 63, rl = tid >> 6;
  const float sc = scale[c], sh = shift[c];
  float s = 0.f;
  int cnt = 0;
  for (int i = 0; i < NTOT / 4; ++i) {
    const int r = rl + 4 * i;
    if (Nseg[r] == g) {
      s += X[(size_t)r * NEMB + c] * sc + sh;
      cnt++;
    }
  }
  __shared__ float ls[4][64];
  __shared__ int lc[4];
  ls[rl][c] = s;
  if (c == 0) lc[rl] = cnt;
  __syncthreads();
  if (tid < 64) {
    const float S = ls[0][tid] + ls[1][tid] + ls[2][tid] + ls[3][tid];
    const float C = (float)(lc[0] + lc[1] + lc[2] + lc[3]);
    pooled[g * 64 + tid] = S / fmaxf(C, 1.f);
  }
}

// ---------------------------------------------------------------------------
// K5: readout MLP: out = relu(pooled@Wn1 + bn1) @ Wn2 + bn2
// ---------------------------------------------------------------------------
__global__ __launch_bounds__(128) void mlp_kernel(
    const float* __restrict__ pooled, const float* __restrict__ Wn1,
    const float* __restrict__ bn1, const float* __restrict__ Wn2,
    const float* __restrict__ bn2, float* __restrict__ out) {
  const int g = blockIdx.x, j = threadIdx.x;
  float h = bn1[j];
  for (int k = 0; k < 64; ++k) h += pooled[g * 64 + k] * Wn1[k * 128 + j];
  h = fmaxf(h, 0.f);
  __shared__ float s[128];
  s[j] = h * Wn2[j];
  __syncthreads();
  for (int off = 64; off > 0; off >>= 1) {
    if (j < off) s[j] += s[j + off];
    __syncthreads();
  }
  if (j == 0) out[g] = s[0] + bn2[0];
}

// ---------------------------------------------------------------------------
extern "C" void kernel_launch(void* const* d_in, const int* in_sizes, int n_in,
                              void* d_out, int out_size, void* d_ws,
                              size_t ws_size, hipStream_t stream) {
  const float* X     = (const float*)d_in[0];
  const float* A     = (const float*)d_in[1];
  // d_in[2]=E, d_in[3]=E_avg: unused by the reference
  const int*   Nseg  = (const int*)d_in[4];
  const float* We    = (const float*)d_in[5];
  const float* be    = (const float*)d_in[6];
  const float* W1    = (const float*)d_in[7];
  const float* b1    = (const float*)d_in[8];
  const float* W2    = (const float*)d_in[9];
  const float* b2    = (const float*)d_in[10];
  const float* gamma = (const float*)d_in[11];
  const float* beta  = (const float*)d_in[12];
  const float* Wn1   = (const float*)d_in[13];
  const float* bn1   = (const float*)d_in[14];
  const float* Wn2   = (const float*)d_in[15];
  const float* bn2   = (const float*)d_in[16];
  float* out = (float*)d_out;

  char* ws = (char*)d_ws;
  float*  X0     = (float*)(ws);                       // 4 MB
  float*  X1     = (float*)(ws + ((size_t)4 << 20));   // 4 MB
  __bf16* Yh     = (__bf16*)(ws + ((size_t)8 << 20));  // 2 MB
  __bf16* Yl     = (__bf16*)(ws + ((size_t)10 << 20)); // 2 MB
  float*  psum   = (float*)(ws + ((size_t)12 << 20));  // 64*64
  float*  psq    = psum + 64 * 64;
  float*  scale  = psq + 64 * 64;
  float*  shiftv = scale + 64;
  float*  pooled = shiftv + 64;                        // 64*64

  embed_kernel<<<NTOT / 4, 256, 0, stream>>>(X, We, be, X0);

  // gc layer 1: X1 = X0 + leaky_relu(A @ (X0@W1) + b1)
  xw_kernel<<<NTOT / 64, 256, 0, stream>>>(X0, W1, Yh, Yl);
  spmm_kernel<<<NTOT / BM, 256, 0, stream>>>(A, Yh, Yl, X0, b1, X1);

  // gc layer 2: X0 = X1 + leaky_relu(A @ (X1@W2) + b2)
  xw_kernel<<<NTOT / 64, 256, 0, stream>>>(X1, W2, Yh, Yl);
  spmm_kernel<<<NTOT / BM, 256, 0, stream>>>(A, Yh, Yl, X1, b2, X0);

  bn_partial<<<64, 256, 0, stream>>>(X0, psum, psq);
  bn_final<<<1, 64, 0, stream>>>(psum, psq, gamma, beta, scale, shiftv);
  pool_kernel<<<64, 256, 0, stream>>>(X0, Nseg, scale, shiftv, pooled);
  mlp_kernel<<<64, 128, 0, stream>>>(pooled, Wn1, bn1, Wn2, bn2, out);
}
